// GraphConv_17532056502697
// MI455X (gfx1250) — matrix operand order repaired and checked by reading, not compiled, not run on hardware
//
#include <hip/hip_runtime.h>

typedef __attribute__((ext_vector_type(16))) _Float16 v16h;
typedef __attribute__((ext_vector_type(8)))  _Float16 v8h;
typedef __attribute__((ext_vector_type(8)))  float    v8f;

#define D_FEAT 64
#define D_EDGE 16
#define D_HID  16
#define D_MSG  64
#define D_OUT  64

// packed-weight fragment regions (in halves) inside d_ws
#define W1P_OFF 0       // msg_W1: 3 K-tiles * 512
#define W2P_OFF 1536    // msg_W2: 4 N-tiles * 512
#define U1P_OFF 3584    // udt_W1: 4 K-tiles * 512
#define U2P_OFF 5632    // udt_W2: 4 N-tiles * 512
#define WP_TOTAL 7680   // halves (15360 B)
#define RKEYS_BYTE_OFF 16384

// ---- monotone float <-> uint32 encoding so segment-max = atomicMax(u32) ----
__device__ __forceinline__ unsigned f32_key(float f) {
  unsigned u = __float_as_uint(f);
  return (u & 0x80000000u) ? ~u : (u | 0x80000000u);
}
// key 0 (memset sentinel) decodes to NaN -> "no in-edge" -> 0 (matches reference)
__device__ __forceinline__ float key_f32_or_zero(unsigned k) {
  unsigned u = (k & 0x80000000u) ? (k & 0x7fffffffu) : ~k;
  if (((u >> 23) & 0xffu) == 0xffu) return 0.0f;
  return __uint_as_float(u);
}

__device__ __forceinline__ v8h cvt8(float4 a, float4 b) {
  v8h h;
  h[0] = (_Float16)a.x; h[1] = (_Float16)a.y; h[2] = (_Float16)a.z; h[3] = (_Float16)a.w;
  h[4] = (_Float16)b.x; h[5] = (_Float16)b.y; h[6] = (_Float16)b.z; h[7] = (_Float16)b.w;
  return h;
}

// Pre-pack all four weight matrices into WMMA B-fragment order (f16, zero padded).
// B layout (16-bit, 16x16x32): lane holds column N=lane&15; halves j=0..15 map to
// K = Ktile*32 + (j>>3)*16 + (lane>>4)*8 + (j&7).  One fragment = 512 halves.
__global__ void pack_weights(const float* __restrict__ mW1, const float* __restrict__ mW2,
                             const float* __restrict__ uW1, const float* __restrict__ uW2,
                             _Float16* __restrict__ wp) {
  const int i = blockIdx.x * blockDim.x + threadIdx.x;
  if (i >= WP_TOTAL) return;
  float val = 0.0f;
  if (i < W2P_OFF) {                       // msg_W1 [80][16], K padded to 96
    const int idx = i, t = idx >> 9, rem = idx & 511, lane = rem >> 4, j = rem & 15;
    const int K = t * 32 + ((j & 8) << 1) + ((lane >> 4) << 3) + (j & 7), N = lane & 15;
    if (K < D_EDGE + D_FEAT) val = mW1[K * D_HID + N];
  } else if (i < U1P_OFF) {                // msg_W2 [16][64], K padded to 32, 4 N-tiles
    const int idx = i - W2P_OFF, nt = idx >> 9, rem = idx & 511, lane = rem >> 4, j = rem & 15;
    const int K = ((j & 8) << 1) + ((lane >> 4) << 3) + (j & 7), N = lane & 15;
    if (K < D_HID) val = mW2[K * D_MSG + nt * 16 + N];
  } else if (i < U2P_OFF) {                // udt_W1 [128][16]
    const int idx = i - U1P_OFF, t = idx >> 9, rem = idx & 511, lane = rem >> 4, j = rem & 15;
    const int K = t * 32 + ((j & 8) << 1) + ((lane >> 4) << 3) + (j & 7), N = lane & 15;
    val = uW1[K * D_HID + N];
  } else {                                 // udt_W2 [16][64]
    const int idx = i - U2P_OFF, nt = idx >> 9, rem = idx & 511, lane = rem >> 4, j = rem & 15;
    const int K = ((j & 8) << 1) + ((lane >> 4) << 3) + (j & 7), N = lane & 15;
    if (K < D_HID) val = uW2[K * D_OUT + nt * 16 + N];
  }
  wp[i] = (_Float16)val;
}

// Build a 16x16x32 A fragment from an LDS row: halves 0..7 = K base..base+7,
// halves 8..15 = K base+16..base+23, base = kt*32 + (lane>>4)*8.
__device__ __forceinline__ v16h a_frag(const _Float16* arow, int kt, int half) {
  const v8h lo = *(const v8h*)(arow + kt * 32 + half * 8);
  const v8h hi = *(const v8h*)(arow + kt * 32 + 16 + half * 8);
  return __builtin_shufflevector(lo, hi, 0, 1, 2, 3, 4, 5, 6, 7,
                                 8, 9, 10, 11, 12, 13, 14, 15);
}

// One wave32 per 16 edges.
__global__ __launch_bounds__(32) void edge_msg_kernel(
    const float* __restrict__ x, const float* __restrict__ ef,
    const int* __restrict__ src, const int* __restrict__ dst,
    const _Float16* __restrict__ wp, const float* __restrict__ b1,
    const float* __restrict__ b2, unsigned* __restrict__ rkeys, int nEdges)
{
  __shared__ __align__(16) _Float16 ain[16 * 96];   // [ef(16)|x(64)|pad(16)] f16
  __shared__ __align__(16) _Float16 hT[16 * 16];
  const int l  = threadIdx.x;
  const int e0 = blockIdx.x * 16;

  // ---- cooperative staging: contiguous float4 loads, packed f16 stores ----
  {
    const int r = l >> 1, ch = l & 1;
    int er = e0 + r; if (er >= nEdges) er = nEdges - 1;
    const float4* ep = (const float4*)(ef + (size_t)er * D_EDGE + ch * 8);
    *(v8h*)&ain[r * 96 + ch * 8] = cvt8(ep[0], ep[1]);
    const int s = src[er];
    const float4* xp = (const float4*)(x + (size_t)s * D_FEAT + ch * 32);
    #pragma unroll
    for (int q = 0; q < 4; ++q)
      *(v8h*)&ain[r * 96 + 16 + ch * 32 + q * 8] = cvt8(xp[2 * q], xp[2 * q + 1]);
    const v8h z = {};
    *(v8h*)&ain[r * 96 + 80 + ch * 8] = z;
  }
  __syncthreads();

  const int N = l & 15, half = l >> 4;
  const _Float16* arow = ain + N * 96;     // A row M = lane&15
  const v16h* W1f = (const v16h*)(wp + W1P_OFF);
  const v16h* W2f = (const v16h*)(wp + W2P_OFF);

  v8f c = {};
  #pragma unroll
  for (int kt = 0; kt < 3; ++kt) {
    const v16h a = a_frag(arow, kt, half);
    const v16h b = W1f[kt * 32 + l];
    c = __builtin_amdgcn_wmma_f32_16x16x32_f16(false, a, false, b, (short)0, c,
                                               false, false);
  }

  {  // bias + relu; C layout: element i -> row M = i + 8*half, col N
    const float bias = b1[N];
    #pragma unroll
    for (int i = 0; i < 8; ++i)
      hT[(i + 8 * half) * 16 + N] = (_Float16)fmaxf(c[i] + bias, 0.0f);
  }
  __syncthreads();

  const v8h hlo = *(const v8h*)(hT + N * 16 + half * 8);
  const v8h z8 = {};
  const v16h a2 = __builtin_shufflevector(hlo, z8, 0, 1, 2, 3, 4, 5, 6, 7,
                                          8, 9, 10, 11, 12, 13, 14, 15);

  int em = e0 + N; if (em >= nEdges) em = nEdges - 1;
  const int dN = dst[em];                  // lane N holds dst of row N

  #pragma unroll
  for (int nt = 0; nt < 4; ++nt) {
    const v16h b2v = W2f[nt * 32 + l];
    v8f c2 = {};
    c2 = __builtin_amdgcn_wmma_f32_16x16x32_f16(false, a2, false, b2v, (short)0,
                                                c2, false, false);
    const float bias2 = b2[nt * 16 + N];
    #pragma unroll
    for (int i = 0; i < 8; ++i) {
      const int dd = __shfl(dN, i + 8 * half, 32);   // dst of row M=i+8*half
      atomicMax(&rkeys[(size_t)dd * D_MSG + (nt * 16 + N)],
                f32_key(c2[i] + bias2));
    }
  }
}

// One wave32 per 16 nodes.
__global__ __launch_bounds__(32) void node_update_kernel(
    const float* __restrict__ x, const unsigned* __restrict__ rkeys,
    const _Float16* __restrict__ wp, const float* __restrict__ b1,
    const float* __restrict__ b2, float* __restrict__ out, int nNodes)
{
  __shared__ __align__(16) _Float16 ain[16 * 128];  // [x(64)|r(64)] f16
  __shared__ __align__(16) _Float16 hT[16 * 16];
  __shared__ __align__(16) float    outT[16 * 64];
  const int l  = threadIdx.x;
  const int v0 = blockIdx.x * 16;

  {
    const int r = l >> 1, ch = l & 1;
    int vr = v0 + r; if (vr >= nNodes) vr = nNodes - 1;
    const float4* xp = (const float4*)(x + (size_t)vr * D_FEAT + ch * 32);
    #pragma unroll
    for (int q = 0; q < 4; ++q)
      *(v8h*)&ain[r * 128 + ch * 32 + q * 8] = cvt8(xp[2 * q], xp[2 * q + 1]);
    const uint4* rp = (const uint4*)(rkeys + (size_t)vr * D_MSG + ch * 32);
    #pragma unroll
    for (int q = 0; q < 4; ++q) {
      const uint4 k0 = rp[2 * q], k1 = rp[2 * q + 1];
      v8h h;
      h[0] = (_Float16)key_f32_or_zero(k0.x); h[1] = (_Float16)key_f32_or_zero(k0.y);
      h[2] = (_Float16)key_f32_or_zero(k0.z); h[3] = (_Float16)key_f32_or_zero(k0.w);
      h[4] = (_Float16)key_f32_or_zero(k1.x); h[5] = (_Float16)key_f32_or_zero(k1.y);
      h[6] = (_Float16)key_f32_or_zero(k1.z); h[7] = (_Float16)key_f32_or_zero(k1.w);
      *(v8h*)&ain[r * 128 + 64 + ch * 32 + q * 8] = h;
    }
  }
  __syncthreads();

  const int N = l & 15, half = l >> 4;
  const _Float16* arow = ain + N * 128;
  const v16h* U1f = (const v16h*)(wp + U1P_OFF);
  const v16h* U2f = (const v16h*)(wp + U2P_OFF);

  v8f c = {};
  #pragma unroll
  for (int kt = 0; kt < 4; ++kt) {
    const v16h a = a_frag(arow, kt, half);
    const v16h b = U1f[kt * 32 + l];
    c = __builtin_amdgcn_wmma_f32_16x16x32_f16(false, a, false, b, (short)0, c,
                                               false, false);
  }

  {
    const float bias = b1[N];
    #pragma unroll
    for (int i = 0; i < 8; ++i)
      hT[(i + 8 * half) * 16 + N] = (_Float16)fmaxf(c[i] + bias, 0.0f);
  }
  __syncthreads();

  const v8h hlo = *(const v8h*)(hT + N * 16 + half * 8);
  const v8h z8 = {};
  const v16h a2 = __builtin_shufflevector(hlo, z8, 0, 1, 2, 3, 4, 5, 6, 7,
                                          8, 9, 10, 11, 12, 13, 14, 15);

  #pragma unroll
  for (int nt = 0; nt < 4; ++nt) {
    const v16h b2v = U2f[nt * 32 + l];
    v8f c2 = {};
    c2 = __builtin_amdgcn_wmma_f32_16x16x32_f16(false, a2, false, b2v, (short)0,
                                                c2, false, false);
    const float bias2 = b2[nt * 16 + N];
    #pragma unroll
    for (int i = 0; i < 8; ++i)
      outT[(i + 8 * half) * 64 + nt * 16 + N] = c2[i] + bias2;
  }
  __syncthreads();

  {  // coalesced store: each lane writes 32 contiguous floats of one row
    const int r = l >> 1, ch = l & 1;
    const int vr = v0 + r;
    if (vr < nNodes) {
      float4* op = (float4*)(out + (size_t)vr * D_OUT + ch * 32);
      const float4* tp = (const float4*)(outT + r * 64 + ch * 32);
      #pragma unroll
      for (int q = 0; q < 8; ++q) op[q] = tp[q];
    }
  }
}

extern "C" void kernel_launch(void* const* d_in, const int* in_sizes, int n_in,
                              void* d_out, int out_size, void* d_ws, size_t ws_size,
                              hipStream_t stream) {
  (void)n_in; (void)out_size; (void)ws_size;
  const float* x      = (const float*)d_in[0];
  const float* ef     = (const float*)d_in[1];
  const int*   src    = (const int*)d_in[2];
  const int*   dst    = (const int*)d_in[3];
  const float* msg_W1 = (const float*)d_in[4];
  const float* msg_b1 = (const float*)d_in[5];
  const float* msg_W2 = (const float*)d_in[6];
  const float* msg_b2 = (const float*)d_in[7];
  const float* udt_W1 = (const float*)d_in[8];
  const float* udt_b1 = (const float*)d_in[9];
  const float* udt_W2 = (const float*)d_in[10];
  const float* udt_b2 = (const float*)d_in[11];

  const int nEdges = in_sizes[2];            // 800000
  const int nNodes = in_sizes[0] / D_FEAT;   // 50000

  _Float16* wp    = (_Float16*)d_ws;
  unsigned* rkeys = (unsigned*)((char*)d_ws + RKEYS_BYTE_OFF);

  pack_weights<<<(WP_TOTAL + 255) / 256, 256, 0, stream>>>(
      msg_W1, msg_W2, udt_W1, udt_W2, wp);

  hipMemsetAsync(rkeys, 0, (size_t)nNodes * D_MSG * sizeof(unsigned), stream);

  edge_msg_kernel<<<(nEdges + 15) / 16, 32, 0, stream>>>(
      x, ef, src, dst, wp, msg_b1, msg_b2, rkeys, nEdges);

  node_update_kernel<<<(nNodes + 15) / 16, 32, 0, stream>>>(
      x, rkeys, wp, udt_b1, udt_b2, (float*)d_out, nNodes);
}
